// YvFineGrainedRouter_81003083203657
// MI455X (gfx1250) — matrix operand order repaired and checked by reading, not compiled, not run on hardware
//
#include <hip/hip_runtime.h>

// ---------------------------------------------------------------------------
// MI455X (gfx1250) UltraMem-style router:
//   logits = x[16384,4096] @ W^T[4096,256]  (bf16 WMMA, f32 accumulate)
//   -> 16x16 grid double-softmax -> outer product -> softmax(256) -> top-8
// GEMM is HBM-bound (268 MB of x @ 23.3 TB/s ~ 11.5us); bf16 WMMA keeps the
// math under the memory floor. W is repacked once into the CDNA5 B-fragment
// layout so every B load is lane-contiguous from L2. A fragments are loaded
// straight from global x (two contiguous 32B fp32 runs per lane match the
// ISA 7.12.2 A layout), packed with v_cvt_pk_bf16_f32 -- no LDS staging, no
// barriers, no accumulator register rotation.
// ---------------------------------------------------------------------------

typedef __bf16 bf16_t;
typedef __attribute__((ext_vector_type(16))) __bf16 v16bf;
typedef __attribute__((ext_vector_type(8)))  float  v8f;
typedef __attribute__((ext_vector_type(4)))  float  v4f;

#define HDIM    4096
#define NEXP    256
#define KCHUNKS (HDIM / 32)

// ---------------------------------------------------------------------------
// Kernel 0: repack W[256,4096] fp32 -> bf16 fragments in WMMA B layout.
// Fragment (kc, nt): B tile is 32(K) x 16(N), per ISA 7.12.2 (wave32):
//   lane l holds N = nt*16 + (l&15), element e holds K = 8*(l>>4) + (e&7) + 16*(e>>3).
// Stored as wfrag[((kc*16+nt)*32 + l)*16 + e]  (2 MiB total, L2-resident).
// ---------------------------------------------------------------------------
__global__ void prep_w(const float* __restrict__ W, bf16_t* __restrict__ wfrag)
{
    int idx = blockIdx.x * blockDim.x + threadIdx.x;   // 0 .. 2^20-1
    int e   = idx & 15;          // element -> K within chunk
    int l   = (idx >> 4) & 31;   // lane -> N within tile (l&15), K half (l>>4)
    int nt  = (idx >> 9) & 15;   // N tile
    int kc  = idx >> 13;         // K chunk
    int n   = nt * 16 + (l & 15);
    int k   = kc * 32 + 8 * (l >> 4) + (e & 7) + 16 * (e >> 3);
    wfrag[idx] = (bf16_t)W[(size_t)n * HDIM + k];
}

// ---------------------------------------------------------------------------
// Kernel 1: GEMM. Block = 256 threads (8 waves) owns 32 rows x 256 experts.
// wave -> (mgroup = wave>>2 owns 16 rows, nquad = wave&3 owns 64 experts).
// The 4 nquad waves sharing a row group re-read the same x cachelines: the
// first touch pulls the 128B line (exactly one K-chunk of one row) into L2,
// the rest hit L2. x leaves HBM exactly once.
// ---------------------------------------------------------------------------
__global__ __launch_bounds__(256, 2)
void gemm_logits(const float* __restrict__ x, const bf16_t* __restrict__ wfrag,
                 float* __restrict__ logits)
{
    const int tid    = threadIdx.x;
    const int wave   = tid >> 5;
    const int lane   = tid & 31;
    const int mgroup = wave >> 2;      // 0..1 : 16-row group
    const int nquad  = wave & 3;       // 0..3 : 64-expert group
    const int h      = lane >> 4;      // lane half
    const size_t mblk = (size_t)blockIdx.x * 32;

    // A-fragment base for this lane: row M = lane&15 of the group, fp32
    // runs start at K offsets 8h and 16+8h within each 32-wide K chunk.
    const size_t mrow = mblk + mgroup * 16 + (lane & 15);
    const float* ap = x + mrow * HDIM + 8 * h;

    v8f zero = {};
    v8f acc[4];
#pragma unroll
    for (int i = 0; i < 4; ++i) acc[i] = zero;

    for (int kc = 0; kc < KCHUNKS; ++kc) {
        // ---- A fragment: 2 contiguous 32B fp32 runs -> 4 global_load_b128,
        // packed pairwise (v_cvt_pk_bf16_f32) into VGPR order: element 2i,
        // 2i+1 live in VGPR i lo/hi, and K is consecutive within each run.
        const float* a_ptr = ap + kc * 32;
        v4f a0 = *(const v4f*)(a_ptr);
        v4f a1 = *(const v4f*)(a_ptr + 4);
        v4f a2 = *(const v4f*)(a_ptr + 16);
        v4f a3 = *(const v4f*)(a_ptr + 20);
        v16bf a;
#pragma unroll
        for (int i = 0; i < 4; ++i) {
            a[i]      = (bf16_t)a0[i];
            a[4 + i]  = (bf16_t)a1[i];
            a[8 + i]  = (bf16_t)a2[i];
            a[12 + i] = (bf16_t)a3[i];
        }

        // ---- 4 B fragments (lane-contiguous 32 B each, L2-hot) + 4 WMMAs
        const v16bf* bbase =
            (const v16bf*)wfrag + ((size_t)kc * 16 + nquad * 4) * 32 + lane;
#pragma unroll
        for (int nt = 0; nt < 4; ++nt) {
            v16bf b = bbase[nt * 32];
            acc[nt] = __builtin_amdgcn_wmma_f32_16x16x32_bf16(
                false, a, false, b, (short)0, acc[nt], false, false);
        }
    }

    // ---- store logits. D layout: VGPR r, lane: N = lane&15, M = r + 8h.
#pragma unroll
    for (int nt = 0; nt < 4; ++nt) {
        const int ncol = nquad * 64 + nt * 16 + (lane & 15);
        float* op = logits + (mblk + mgroup * 16 + h * 8) * NEXP + ncol;
#pragma unroll
        for (int r = 0; r < 8; ++r) op[(size_t)r * NEXP] = acc[nt][r];
    }
}

// ---------------------------------------------------------------------------
// Kernel 2: routing epilogue. 16 lanes per token; lane r owns grid row r.
// ---------------------------------------------------------------------------
__global__ __launch_bounds__(256)
void router_epilogue(const float* __restrict__ logits, float* __restrict__ out,
                     int ntok)
{
    const int tid   = threadIdx.x;
    const int r     = tid & 15;                 // grid row owned by this lane
    const int token = blockIdx.x * 16 + (tid >> 4);
    if (token >= ntok) return;                  // whole 16-lane group exits

    const float4* lg = (const float4*)(logits + (size_t)token * 256 + r * 16);
    float4 q0 = lg[0], q1 = lg[1], q2 = lg[2], q3 = lg[3];
    float v[16] = { q0.x, q0.y, q0.z, q0.w,  q1.x, q1.y, q1.z, q1.w,
                    q2.x, q2.y, q2.z, q2.w,  q3.x, q3.y, q3.z, q3.w };

    // row softmax -> mean over same axis (numerically, not symbolically)
    float m = v[0];
#pragma unroll
    for (int c = 1; c < 16; ++c) m = fmaxf(m, v[c]);
    float e[16], s = 0.f;
#pragma unroll
    for (int c = 0; c < 16; ++c) { e[c] = __expf(v[c] - m); s += e[c]; }
    float inv_s = 1.f / s, rscore = 0.f;
#pragma unroll
    for (int c = 0; c < 16; ++c) rscore += e[c] * inv_s;
    rscore *= (1.f / 16.f);

    // column softmax -> mean (cross-lane butterflies stay within 16-group)
    float w[16];
    float gmax = -__builtin_inff();
#pragma unroll
    for (int c = 0; c < 16; ++c) {
        float mc = v[c];
#pragma unroll
        for (int off = 8; off; off >>= 1) mc = fmaxf(mc, __shfl_xor(mc, off));
        float ec = __expf(v[c] - mc);
        float sc = ec;
#pragma unroll
        for (int off = 8; off; off >>= 1) sc += __shfl_xor(sc, off);
        float cs = ec / sc;
#pragma unroll
        for (int off = 8; off; off >>= 1) cs += __shfl_xor(cs, off);
        float lg2 = rscore * (cs * (1.f / 16.f));   // outer product entry
        w[c] = lg2;
        gmax = fmaxf(gmax, lg2);
    }

    // softmax over all 256
#pragma unroll
    for (int off = 8; off; off >>= 1) gmax = fmaxf(gmax, __shfl_xor(gmax, off));
    float gs = 0.f;
#pragma unroll
    for (int c = 0; c < 16; ++c) { w[c] = __expf(w[c] - gmax); gs += w[c]; }
#pragma unroll
    for (int off = 8; off; off >>= 1) gs += __shfl_xor(gs, off);
    float ginv = 1.f / gs;
#pragma unroll
    for (int c = 0; c < 16; ++c) w[c] *= ginv;

    // stable top-8 (ties -> lowest index, matching jax.lax.top_k)
    float topv[8]; int topi[8];
#pragma unroll
    for (int t = 0; t < 8; ++t) {
        float bv = -__builtin_inff(); int bi = 0x7fffffff;
#pragma unroll
        for (int c = 0; c < 16; ++c) {
            int idx = r * 16 + c;
            if (w[c] > bv || (w[c] == bv && idx < bi)) { bv = w[c]; bi = idx; }
        }
#pragma unroll
        for (int off = 8; off; off >>= 1) {
            float ov = __shfl_xor(bv, off);
            int   oi = __shfl_xor(bi, off);
            if (ov > bv || (ov == bv && oi < bi)) { bv = ov; bi = oi; }
        }
        topv[t] = bv; topi[t] = bi;
        if ((bi >> 4) == r) w[bi & 15] = -__builtin_inff();
    }

    float tsum = 0.f;
#pragma unroll
    for (int t = 0; t < 8; ++t) tsum += topv[t];
    float tnorm = 1.f / (tsum + 1e-8f);

    // outputs concatenated flat: top_w | expert_idx | sub_idx | loss
    if (r < 8) {
        size_t base = (size_t)token * 8 + r;
        out[base]                        = topv[r] * tnorm;
        out[(size_t)ntok * 8  + base]    = (float)(topi[r] >> 2);  // idx // 4
        out[(size_t)ntok * 16 + base]    = (float)(topi[r] & 3);   // idx % 4
    }
    if (token == 0 && tid == 0) out[(size_t)ntok * 24] = 0.0f;     // aux loss
}

// ---------------------------------------------------------------------------
extern "C" void kernel_launch(void* const* d_in, const int* in_sizes, int n_in,
                              void* d_out, int out_size, void* d_ws, size_t ws_size,
                              hipStream_t stream)
{
    (void)n_in; (void)out_size; (void)ws_size;
    const float* x = (const float*)d_in[0];
    const float* W = (const float*)d_in[1];
    float* out = (float*)d_out;
    const int ntok = in_sizes[0] / HDIM;     // 16384

    bf16_t* wfrag = (bf16_t*)d_ws;                                        // 2 MiB
    float*  logits = (float*)((char*)d_ws +
                              (size_t)NEXP * HDIM * sizeof(bf16_t));      // 16 MiB

    prep_w<<<(NEXP * HDIM) / 256, 256, 0, stream>>>(W, wfrag);
    gemm_logits<<<ntok / 32, 256, 0, stream>>>(x, wfrag, logits);
    router_epilogue<<<(ntok + 15) / 16, 256, 0, stream>>>(logits, out, ntok);
}
